// FNO2dOTP_48275432407397
// MI455X (gfx1250) — matrix operand order repaired and checked by reading, not compiled, not run on hardware
//
#include <hip/hip_runtime.h>
#include <hip/hip_bf16.h>
#include <math.h>

// ---------------------------------------------------------------------------
// FNO2d forward on gfx1250 (MI455X), expressed entirely as batched WMMA GEMMs.
// rfft2/irfft2 with mode truncation == low-rank DFT matrices -> GEMM chains.
// LDS-staged tiles: wide global_load_b128 -> LDS (f16) -> ds_load fragments.
// fc1+fc2 fused (kills the 134MB F1 round-trip).
// ---------------------------------------------------------------------------

typedef __attribute__((ext_vector_type(16))) _Float16 v16h;
typedef __attribute__((ext_vector_type(8)))  _Float16 v8h;
typedef __attribute__((ext_vector_type(8)))  float    v8f;

#define BD   16
#define CINC 200
#define CH   64
#define HH   128
#define WW   128
#define NPIX (HH * WW)
#define MM1  16
#define MM2  16

#define SH_STRIDE 48  // halves per LDS tile row (96 B: keeps fragments 16/32-B aligned)

// ---- workspace layout (float element offsets) ----
static const long long OFF_FW   = 0;                       // [128][32] fwd-W basis (cos | -sin)
static const long long OFF_FHR  = OFF_FW   + 128 * 32;     // [32][128] fwd-H real
static const long long OFF_FHI  = OFF_FHR  + 32 * 128;     // [32][128] fwd-H imag (-sin)
static const long long OFF_GHR  = OFF_FHI  + 32 * 128;     // [128][32] inv-H real (1/H)
static const long long OFF_GHI  = OFF_GHR  + 128 * 32;     // [128][32] inv-H imag (1/H)
static const long long OFF_GC   = OFF_GHI  + 128 * 32;     // [16][128] inv-W cos (doubling, 1/W)
static const long long OFF_GS   = OFF_GC   + 16 * 128;     // [16][128] inv-W -sin
static const long long OFF_ACT0 = OFF_GS   + 16 * 128;
static const long long OFF_ACT1 = OFF_ACT0 + (long long)BD * CH * NPIX;
static const long long OFF_U    = OFF_ACT1 + (long long)BD * CH * NPIX;   // [b,c][128][32]
static const long long OFF_V    = OFF_U    + (long long)BD * CH * 128*32; // [b,c][2][512]
static const long long OFF_O    = OFF_V    + (long long)BD * CH * 1024;   // [b,c][2][512]
static const long long OFF_Z    = OFF_O    + (long long)BD * CH * 1024;   // [b,c][2][2048]
static const long long OFF_T    = OFF_Z    + (long long)BD * CH * 4096;   // [b,c][16384]

struct GemmArgs {
  const float* A;  long long bsA;  int rsA, csA;
  const float* A2; long long bsA2; int rsA2, csA2;
  const float* B;  long long bsB;  int rsB, csB;
  const float* B2; long long bsB2; int rsB2, csB2;
  const float* S;  long long bsS;  int rsS, csS;
  const float* bias;
  float* C;        long long bsC;  int rsC, csC;
  int M, N, K;
  float s2;        // sign applied to A2*B2 product (folded into A2 staging)
  int biasMode;    // 0 none, 1 bias[row], 2 bias[bz & (biasMod-1)]  (biasMod pow2)
  int biasMod;
  int gelu;
  int useS;
  int dual;
};

__device__ __forceinline__ float gelu_exact(float x) {
  return 0.5f * x * (1.0f + erff(x * 0.70710678118654752f));
}

// Stage a 32x32 f32 tile (rows r0.., cols c0..) into LDS as f16.
// trans=false: sh[row][col] (packed b32 stores); trans=true: sh[col][row].
// Requires cLim % 8 == 0 (tile-chunk granular zero padding).
__device__ __forceinline__ void stage32(const float* __restrict__ P,
                                        long long rs, long long cs,
                                        int r0, int rLim, int c0, int cLim,
                                        _Float16* sh, bool trans, float scale,
                                        int t, long long poff) {
  const int row = t >> 2;          // 0..31
  const int cq  = (t & 3) * 8;     // 0,8,16,24
  const bool ok = (r0 + row < rLim) && (c0 + cq < cLim);
  float v[8];
  if (ok && cs == 1) {
    const float* p = P + (long long)(r0 + row) * rs + (c0 + cq);
    float4 x0 = *(const float4*)p;
    float4 x1 = *(const float4*)(p + 4);
    v[0] = x0.x; v[1] = x0.y; v[2] = x0.z; v[3] = x0.w;
    v[4] = x1.x; v[5] = x1.y; v[6] = x1.z; v[7] = x1.w;
    __builtin_prefetch(p + poff, 0, 1);   // speculative: drop-safe (ISA 10.5)
  } else if (ok) {
#pragma unroll
    for (int i = 0; i < 8; ++i)
      v[i] = P[(long long)(r0 + row) * rs + (long long)(c0 + cq + i) * cs];
  } else {
#pragma unroll
    for (int i = 0; i < 8; ++i) v[i] = 0.f;
  }
  if (trans) {
#pragma unroll
    for (int i = 0; i < 8; ++i)
      sh[(cq + i) * SH_STRIDE + row] = (_Float16)(v[i] * scale);
  } else {
#pragma unroll
    for (int i = 0; i < 8; i += 2) {
      union { _Float16 h[2]; unsigned u; } pk;
      pk.h[0] = (_Float16)(v[i] * scale);
      pk.h[1] = (_Float16)(v[i + 1] * scale);
      *(unsigned*)&sh[row * SH_STRIDE + cq + i] = pk.u;  // 4B-aligned (cq even)
    }
  }
}

// Generic batched GEMM: C = A*B [+ s2*A2*B2] [+ S] [+ bias] [gelu].
// f32 memory, f16 WMMA with f32 accumulation. 128 threads = 4 waves,
// 32x32 block tile, one 16x16 WMMA tile per wave.
__global__ __launch_bounds__(128) void wmma_gemm_f16(GemmArgs g) {
  __shared__ __align__(32) _Float16 ashm[32 * SH_STRIDE];
  __shared__ __align__(32) _Float16 bshm[32 * SH_STRIDE];
  __shared__ __align__(32) _Float16 ashm2[32 * SH_STRIDE];
  __shared__ __align__(32) _Float16 bshm2[32 * SH_STRIDE];

  const int t    = threadIdx.x;
  const int lane = t & 31;
  const int wave = t >> 5;
  const int bz   = blockIdx.z;
  const int tMb  = blockIdx.y * 32;           // block tile row base
  const int tNb  = blockIdx.x * 32;           // block tile col base
  const int tMw  = tMb + (wave >> 1) * 16;    // wave tile
  const int tNw  = tNb + (wave & 1) * 16;
  const bool waveLive = (tMw < g.M) && (tNw < g.N);  // wave-uniform

  const float* Ap  = g.A + (long long)bz * g.bsA;
  const float* Bp  = g.B + (long long)bz * g.bsB;
  const float* A2p = g.dual ? (g.A2 + (long long)bz * g.bsA2) : Ap;
  const float* B2p = g.dual ? (g.B2 + (long long)bz * g.bsB2) : Bp;

  // Lane layouts used here (wave32, contiguous k; A and B match):
  // A frag: M = lane&15; K = (lane>=16?16:0) + j
  // B frag: N = lane&15; K = (lane>=16?16:0) + j
  const int lm  = (wave >> 1) * 16 + (lane & 15);  // LDS row for A frag
  const int aKB = (lane >> 4) * 16;
  const int ln  = (wave & 1) * 16 + (lane & 15);   // LDS row for B^T frag
  const int bKB = (lane >> 4) * 16;

  v8f acc = {0.f, 0.f, 0.f, 0.f, 0.f, 0.f, 0.f, 0.f};
  union V16 { _Float16 h[16]; v16h v; };

  for (int k0 = 0; k0 < g.K; k0 += 32) {
    // ---- cooperative staging (all 128 threads) ----
    stage32(Ap, g.rsA, g.csA, tMb, g.M, k0, g.K, ashm, false, 1.f, t, 32);
    stage32(Bp, g.rsB, g.csB, k0, g.K, tNb, g.N, bshm, true, 1.f, t,
            32 * (long long)g.rsB);
    if (g.dual) {
      stage32(A2p, g.rsA2, g.csA2, tMb, g.M, k0, g.K, ashm2, false, g.s2, t, 32);
      stage32(B2p, g.rsB2, g.csB2, k0, g.K, tNb, g.N, bshm2, true, 1.f, t,
              32 * (long long)g.rsB2);
    }
    __syncthreads();

    // ---- WMMA (wave-uniform guard; EXEC all-ones inside) ----
    if (waveLive) {
      V16 a, b;
      v8h lo = *(const v8h*)&ashm[lm * SH_STRIDE + aKB];
      v8h hi = *(const v8h*)&ashm[lm * SH_STRIDE + aKB + 8];
#pragma unroll
      for (int j = 0; j < 8; ++j) { a.h[j] = lo[j]; a.h[j + 8] = hi[j]; }
      b.v = *(const v16h*)&bshm[ln * SH_STRIDE + bKB];
      acc = __builtin_amdgcn_wmma_f32_16x16x32_f16(false, a.v, false, b.v,
                                                   (short)0, acc, false, false);
      if (g.dual) {
        V16 a2, b2;
        v8h lo2 = *(const v8h*)&ashm2[lm * SH_STRIDE + aKB];
        v8h hi2 = *(const v8h*)&ashm2[lm * SH_STRIDE + aKB + 8];
#pragma unroll
        for (int j = 0; j < 8; ++j) { a2.h[j] = lo2[j]; a2.h[j + 8] = hi2[j]; }
        b2.v = *(const v16h*)&bshm2[ln * SH_STRIDE + bKB];
        acc = __builtin_amdgcn_wmma_f32_16x16x32_f16(false, a2.v, false, b2.v,
                                                     (short)0, acc, false, false);
      }
    }
    __syncthreads();
  }

  if (!waveLive) return;

  // C/D layout: VGPR v holds M = v + (lane>=16)*8, N = lane&15
  const int rB = (lane >> 4) * 8;
  const int nC = tNw + (lane & 15);
#pragma unroll
  for (int v = 0; v < 8; ++v) {
    const int r = tMw + rB + v;
    if (r >= g.M || nC >= g.N) continue;
    float x = acc[v];
    if (g.useS)
      x += g.S[(long long)bz * g.bsS + (long long)r * g.rsS + (long long)nC * g.csS];
    if (g.biasMode == 1)      x += g.bias[r];
    else if (g.biasMode == 2) x += g.bias[bz & (g.biasMod - 1)];
    if (g.gelu) x = gelu_exact(x);
    g.C[(long long)bz * g.bsC + (long long)r * g.rsC + (long long)nC * g.csC] = x;
  }
}

// ---------------------------------------------------------------------------
// Fused fc1 (64->128, bias, GELU) + fc2 (128->1 dot).
// One block = 32 pixels. fc1_w staged fully in LDS (f16), act tile staged,
// 4 co-passes x 2 K-steps of WMMA; fc2 dot accumulated per lane; deterministic
// fixed-slot LDS reduction (no atomics => bit-stable across replays).
// ---------------------------------------------------------------------------
#define FST 80  // LDS row stride in halves (160B, keeps fragment runs 16/32-B aligned)

__global__ __launch_bounds__(128) void fc1fc2_fused(
    const float* __restrict__ act,   // [B][64][NPIX]
    const float* __restrict__ w1,    // [128][64]
    const float* __restrict__ b1,    // [128]
    const float* __restrict__ w2,    // [1][128]
    const float* __restrict__ b2,    // [1]
    float* __restrict__ out)         // [B][NPIX]
{
  __shared__ __align__(32) _Float16 wsh[128 * FST];  // fc1_w [co][k]
  __shared__ __align__(32) _Float16 xsh[32 * FST];   // act^T [pix][k]
  __shared__ float red[128];

  const int t    = threadIdx.x;
  const int lane = t & 31;
  const int wave = t >> 5;
  const int bb   = blockIdx.y;
  const int pix0 = blockIdx.x * 32;

  // ---- stage fc1_w: thread t loads row t (64 floats) ----
  {
    const float* p = w1 + (long long)t * CH;
#pragma unroll
    for (int q = 0; q < 16; ++q) {
      float4 x = ((const float4*)p)[q];
      union { _Float16 h[2]; unsigned u; } p0, p1;
      p0.h[0] = (_Float16)x.x; p0.h[1] = (_Float16)x.y;
      p1.h[0] = (_Float16)x.z; p1.h[1] = (_Float16)x.w;
      *(unsigned*)&wsh[t * FST + q * 4]     = p0.u;
      *(unsigned*)&wsh[t * FST + q * 4 + 2] = p1.u;
    }
  }
  // ---- stage act tile transposed: thread t -> ci = t>>1, 16 pixels ----
  {
    const int ci = t >> 1;
    const int ph = (t & 1) * 16;
    const float* p = act + (long long)bb * CH * NPIX + (long long)ci * NPIX + pix0 + ph;
#pragma unroll
    for (int q = 0; q < 4; ++q) {
      float4 x = ((const float4*)p)[q];
      xsh[(ph + q * 4 + 0) * FST + ci] = (_Float16)x.x;
      xsh[(ph + q * 4 + 1) * FST + ci] = (_Float16)x.y;
      xsh[(ph + q * 4 + 2) * FST + ci] = (_Float16)x.z;
      xsh[(ph + q * 4 + 3) * FST + ci] = (_Float16)x.w;
    }
  }
  __syncthreads();

  const int aKB = (lane >> 4) * 16;
  const int bKB = (lane >> 4) * 16;
  const int pn  = (wave & 1) * 16 + (lane & 15);   // pixel row in xsh
  union V16 { _Float16 h[16]; v16h v; };

  float partial = 0.f;
#pragma unroll
  for (int pass = 0; pass < 4; ++pass) {
    const int tM = pass * 32 + (wave >> 1) * 16;   // co tile base
    const int lm = tM + (lane & 15);
    v8f acc = {0.f, 0.f, 0.f, 0.f, 0.f, 0.f, 0.f, 0.f};
#pragma unroll
    for (int k0 = 0; k0 < CH; k0 += 32) {
      V16 a, b;
      v8h lo = *(const v8h*)&wsh[lm * FST + k0 + aKB];
      v8h hi = *(const v8h*)&wsh[lm * FST + k0 + aKB + 8];
#pragma unroll
      for (int j = 0; j < 8; ++j) { a.h[j] = lo[j]; a.h[j + 8] = hi[j]; }
      b.v = *(const v16h*)&xsh[pn * FST + k0 + bKB];
      acc = __builtin_amdgcn_wmma_f32_16x16x32_f16(false, a.v, false, b.v,
                                                   (short)0, acc, false, false);
    }
    const int rB = (lane >> 4) * 8;
#pragma unroll
    for (int v = 0; v < 8; ++v) {
      const int co = tM + rB + v;
      float y = gelu_exact(acc[v] + b1[co]);
      partial += w2[co] * y;
    }
  }

  // deterministic reduction: fixed slot per (wave,lane)
  red[wave * 32 + lane] = partial;
  __syncthreads();
  if (t < 32) {
    const int wA = (t >> 4) & 1;        // waves covering this pixel half
    const int l0 = t & 15;
    float s = red[wA * 32 + l0]       + red[wA * 32 + l0 + 16]
            + red[(wA + 2) * 32 + l0] + red[(wA + 2) * 32 + l0 + 16];
    out[(long long)bb * NPIX + pix0 + t] = s + b2[0];
  }
}

// DFT basis tables (deterministic, rebuilt each call; a few KB).
__global__ void basis_init(float* ws) {
  const float PI2 = 6.2831853071795864769f;
  const int stride = gridDim.x * blockDim.x;
  const int t = blockIdx.x * blockDim.x + threadIdx.x;
  // forward-W: U[h,k] = sum_w X[h,w] * exp(-2pi i w k / 128), k = 0..15
  for (int i = t; i < 128 * 16; i += stride) {
    int w = i >> 4, k = i & 15;
    float ang = PI2 * (float)((w * k) & 127) / 128.f;
    ws[OFF_FW + (long long)w * 32 + k]      =  cosf(ang);
    ws[OFF_FW + (long long)w * 32 + 16 + k] = -sinf(ang);
  }
  // forward-H over retained rows kh in {0..15, 112..127};
  // inverse-H (ifft along H with 1/H factor): exp(+2pi i kh h / H) / 128
  for (int i = t; i < 32 * 128; i += stride) {
    int kidx = i >> 7, h = i & 127;
    int kh = (kidx < 16) ? kidx : (kidx + 96);
    float ang = PI2 * (float)((kh * h) & 127) / 128.f;
    ws[OFF_FHR + (long long)kidx * 128 + h] =  cosf(ang);
    ws[OFF_FHI + (long long)kidx * 128 + h] = -sinf(ang);
    ws[OFF_GHR + (long long)h * 32 + kidx] = cosf(ang) / 128.f;
    ws[OFF_GHI + (long long)h * 32 + kidx] = sinf(ang) / 128.f;
  }
  // inverse-W (irfft along W, Hermitian doubling; Im of DC dropped):
  // x[h,w] = sum_k zr*Gc[k,w] + zi*Gs[k,w]
  for (int i = t; i < 16 * 128; i += stride) {
    int k = i >> 7, w = i & 127;
    float ang = PI2 * (float)((k * w) & 127) / 128.f;
    float sc = ((k == 0) ? 1.f : 2.f) / 128.f;
    ws[OFF_GC + (long long)k * 128 + w] = sc * cosf(ang);
    ws[OFF_GS + (long long)k * 128 + w] = (k == 0) ? 0.f : (-2.f / 128.f) * sinf(ang);
  }
}

static void launch_gemm(const GemmArgs& g, int batch, hipStream_t s) {
  dim3 grid((unsigned)((g.N + 31) / 32), (unsigned)((g.M + 31) / 32), (unsigned)batch);
  hipLaunchKernelGGL(wmma_gemm_f16, grid, dim3(128), 0, s, g);
}

extern "C" void kernel_launch(void* const* d_in, const int* in_sizes, int n_in,
                              void* d_out, int out_size, void* d_ws, size_t ws_size,
                              hipStream_t stream) {
  const float* x     = (const float*)d_in[0];
  const float* fc0_w = (const float*)d_in[1];
  const float* fc0_b = (const float*)d_in[2];
  const float* w1r   = (const float*)d_in[3];
  const float* w1i   = (const float*)d_in[4];
  const float* w2r   = (const float*)d_in[5];
  const float* w2i   = (const float*)d_in[6];
  const float* wsw   = (const float*)d_in[7];
  const float* wsb   = (const float*)d_in[8];
  const float* fc1_w = (const float*)d_in[9];
  const float* fc1_b = (const float*)d_in[10];
  const float* fc2_w = (const float*)d_in[11];
  const float* fc2_b = (const float*)d_in[12];
  float* ws  = (float*)d_ws;
  float* out = (float*)d_out;

  basis_init<<<64, 256, 0, stream>>>(ws);

  float* ACT[2] = { ws + OFF_ACT0, ws + OFF_ACT1 };
  float* U = ws + OFF_U;
  float* V = ws + OFF_V;
  float* O = ws + OFF_O;
  float* Z = ws + OFF_Z;
  float* T = ws + OFF_T;

  // ---- fc0: [64x200] x [200x16384] per batch ----
  { GemmArgs g{};
    g.A = fc0_w; g.bsA = 0; g.rsA = CINC; g.csA = 1;
    g.B = x;     g.bsB = (long long)CINC * NPIX; g.rsB = NPIX; g.csB = 1;
    g.C = ACT[0]; g.bsC = (long long)CH * NPIX; g.rsC = NPIX; g.csC = 1;
    g.M = CH; g.N = NPIX; g.K = CINC;
    g.bias = fc0_b; g.biasMode = 1;
    launch_gemm(g, BD, stream); }

  int cur = 0;
  for (int l = 0; l < 3; ++l) {
    const float* in = ACT[cur];
    float* outA = ACT[1 - cur];

    // ---- forward DFT along W: [128x128] x [128x32] per (b,c) ----
    { GemmArgs g{};
      g.A = in;            g.bsA = NPIX;    g.rsA = WW; g.csA = 1;
      g.B = ws + OFF_FW;   g.bsB = 0;       g.rsB = 32; g.csB = 1;
      g.C = U;             g.bsC = 128*32;  g.rsC = 32; g.csC = 1;
      g.M = HH; g.N = 32; g.K = WW;
      launch_gemm(g, BD * CH, stream); }

    // ---- forward DFT along H (complex): V = Fh * U ----
    for (int im = 0; im < 2; ++im) {
      GemmArgs g{}; g.dual = 1;
      g.A  = ws + OFF_FHR; g.bsA  = 0; g.rsA  = 128; g.csA  = 1;
      g.A2 = ws + OFF_FHI; g.bsA2 = 0; g.rsA2 = 128; g.csA2 = 1;
      g.B  = U + (im ? 16 : 0); g.bsB  = 128*32; g.rsB  = 32; g.csB  = 1;
      g.B2 = U + (im ? 0 : 16); g.bsB2 = 128*32; g.rsB2 = 32; g.csB2 = 1;
      g.s2 = im ? 1.f : -1.f;
      g.C = V + (long long)im * 512; g.bsC = 1024; g.rsC = MM2; g.csC = 1;
      g.M = 2 * MM1; g.N = MM2; g.K = 128;
      launch_gemm(g, BD * CH, stream);
    }

    // ---- per-mode complex channel mix: O[b,co] = V[b,ci] * W[ci,co] ----
    for (int grp = 0; grp < 2; ++grp) {
      const float* wr = (grp ? w2r : w1r) + (long long)l * CH * CH * MM1 * MM2;
      const float* wi = (grp ? w2i : w1i) + (long long)l * CH * CH * MM1 * MM2;
      for (int im = 0; im < 2; ++im) {
        GemmArgs g{}; g.dual = 1;
        g.A  = V + grp * 256;       g.bsA  = 1; g.rsA  = CH * 1024; g.csA  = 1024;
        g.A2 = V + 512 + grp * 256; g.bsA2 = 1; g.rsA2 = CH * 1024; g.csA2 = 1024;
        g.B  = im ? wi : wr; g.bsB  = 1; g.rsB  = CH * MM1 * MM2; g.csB  = MM1 * MM2;
        g.B2 = im ? wr : wi; g.bsB2 = 1; g.rsB2 = CH * MM1 * MM2; g.csB2 = MM1 * MM2;
        g.s2 = im ? 1.f : -1.f;
        g.C = O + (long long)im * 512 + grp * 256;
        g.bsC = 1; g.rsC = CH * 1024; g.csC = 1024;
        g.M = BD; g.N = CH; g.K = CH;
        launch_gemm(g, MM1 * MM2, stream);  // 256 modes per group
      }
    }

    // ---- inverse DFT along H (complex): Z = Gh * O ----
    for (int im = 0; im < 2; ++im) {
      GemmArgs g{}; g.dual = 1;
      g.A  = ws + OFF_GHR; g.bsA  = 0; g.rsA  = 32; g.csA  = 1;
      g.A2 = ws + OFF_GHI; g.bsA2 = 0; g.rsA2 = 32; g.csA2 = 1;
      g.B  = O + (im ? 512 : 0); g.bsB  = 1024; g.rsB  = 16; g.csB  = 1;
      g.B2 = O + (im ? 0 : 512); g.bsB2 = 1024; g.rsB2 = 16; g.csB2 = 1;
      g.s2 = im ? 1.f : -1.f;
      g.C = Z + (long long)im * 2048; g.bsC = 4096; g.rsC = 16; g.csC = 1;
      g.M = HH; g.N = MM2; g.K = 32;
      launch_gemm(g, BD * CH, stream);
    }

    // ---- ws pointwise path: T = ws_w[l] * act ----
    { GemmArgs g{};
      g.A = wsw + (long long)l * CH * CH; g.bsA = 0; g.rsA = CH; g.csA = 1;
      g.B = in; g.bsB = (long long)CH * NPIX; g.rsB = NPIX; g.csB = 1;
      g.C = T;  g.bsC = (long long)CH * NPIX; g.rsC = NPIX; g.csC = 1;
      g.M = CH; g.N = NPIX; g.K = CH;
      launch_gemm(g, BD, stream); }

    // ---- inverse DFT along W (real out) + T + ws_b, GELU ----
    { GemmArgs g{}; g.dual = 1; g.s2 = 1.f;
      g.A  = Z;        g.bsA  = 4096; g.rsA  = 16; g.csA  = 1;
      g.A2 = Z + 2048; g.bsA2 = 4096; g.rsA2 = 16; g.csA2 = 1;
      g.B  = ws + OFF_GC; g.bsB  = 0; g.rsB  = WW; g.csB  = 1;
      g.B2 = ws + OFF_GS; g.bsB2 = 0; g.rsB2 = WW; g.csB2 = 1;
      g.S = T; g.bsS = NPIX; g.rsS = WW; g.csS = 1; g.useS = 1;
      g.bias = wsb + (long long)l * CH; g.biasMode = 2; g.biasMod = CH;
      g.gelu = 1;
      g.C = outA; g.bsC = NPIX; g.rsC = WW; g.csC = 1;
      g.M = HH; g.N = WW; g.K = MM2;
      launch_gemm(g, BD * CH, stream); }

    cur = 1 - cur;
  }

  // ---- fused fc1 (GELU) + fc2 ----
  fc1fc2_fused<<<dim3(NPIX / 32, BD), 128, 0, stream>>>(
      ACT[cur], fc1_w, fc1_b, fc2_w, fc2_b, out);
}
